// NeuralODEFlow_40269613368069
// MI455X (gfx1250) — matrix operand order repaired and checked by reading, not compiled, not run on hardware
//
#include <hip/hip_runtime.h>
#include <math.h>

#define DIMN   8
#define HID    128
#define BATCHN 4096
#define NSTEPS 3
#define LOG2PI_F 1.8378770664093453f

#define AS3 __attribute__((address_space(3)))

typedef __attribute__((ext_vector_type(16))) __bf16 v16bf;
typedef __attribute__((ext_vector_type(8)))  __bf16 v8bf;
typedef __attribute__((ext_vector_type(8)))  float  v8f;

enum { EMBW=0, EMBB, W00, B00, W01, B01, W10, B10, W11, B11, FINW, FINB };
struct PSet { const float* p[12]; };

// ---------------- LDS layout (byte offsets) ----------------
#define OFF_WT     0        // 4 * 128x128 bf16, W^T  (forward)   = 131072
#define OFF_WB     131072   // 4 * 128x128 bf16, W    (backward)  = 131072
#define OFF_EMBWT  262144   // 128x16 bf16  embW^T (k padded to 16, zeros >=9)
#define OFF_EMBWB  266240   // 16x128 bf16  embW   (rows >=9 zero)
#define OFF_FINWT  270336   // 16x128 bf16  finW^T (rows >=8 zero)
#define OFF_FINWB  274432   // 128x16 bf16  finW   (cols >=8 zero)
#define OFF_EMBB   278528   // 128 f32
#define OFF_BL     279040   // 4*128 f32
#define OFF_FINB   281088   // 8 f32
#define OFF_T1     281216   // t1..t5: 5 * 16x128 bf16 (row-major [m][feat])
#define OFF_GA     (OFF_T1 + 5*4096)
#define OFF_GB     (OFF_GA + 4096)
#define OFF_EPSB   (OFF_GB + 4096)      // 16x16 bf16 (cols >=8 zero)
#define LDS_BYTES  (OFF_EPSB + 512)     // 310400 < 320KB/WGP

struct Ctx {
  const AS3 __bf16 *wt, *wb, *embWT, *embWB, *finWT, *finWB;
  const AS3 float  *embB, *bL, *finB;
  AS3 __bf16 *t1, *t2, *t3, *t4, *t5, *gA, *gB, *epsB;
};

__device__ inline v8f wmma_bf16(v16bf a, v16bf b, v8f c) {
  return __builtin_amdgcn_wmma_f32_16x16x32_bf16(false, a, false, b, (short)0, c, false, false);
}

// Branch-free tanh: tanh(x) = 1 - 2/(exp(2x)+1); exp via v_exp_f32, div via v_rcp_f32.
// Clamp keeps exp finite; accuracy ~1e-7 rel, far beyond bf16 needs.
__device__ inline float fast_tanh(float x) {
  float xc = fminf(15.0f, fmaxf(-15.0f, x));
  float e  = __builtin_amdgcn_exp2f(xc * 2.885390081777927f);   // exp(2*xc)
  float r  = __builtin_amdgcn_rcpf(e + 1.0f);
  return 1.0f - 2.0f * r;
}

__device__ inline v8bf ldsv8(const AS3 __bf16* p) { return *(const AS3 v8bf*)p; }
__device__ inline void stsv8(AS3 __bf16* p, v8bf v) { *(AS3 v8bf*)p = v; }
__device__ inline v8bf zero8() {
  v8bf z;
#pragma unroll
  for (int j = 0; j < 8; ++j) z[j] = (__bf16)0.0f;
  return z;
}
__device__ inline v16bf combine(v8bf lo, v8bf hi) {
  v16bf a;
#pragma unroll
  for (int j = 0; j < 8; ++j) { a[j] = lo[j]; a[j + 8] = hi[j]; }
  return a;
}
__device__ inline v8bf add8(v8bf a, v8bf b) {
  v8bf o;
#pragma unroll
  for (int j = 0; j < 8; ++j) o[j] = (__bf16)((float)a[j] + (float)b[j]);
  return o;
}
__device__ inline v8bf add8_3(v8bf a, v8bf b, v8bf c) {
  v8bf o;
#pragma unroll
  for (int j = 0; j < 8; ++j) o[j] = (__bf16)((float)a[j] + (float)b[j] + (float)c[j]);
  return o;
}

// A fragment (16x32, 16-bit): lane holds row rbase+(lane&15); per half-wave g the K
// chunks are {k0+16g .. +7} and {k0+16g+8 .. +7} -> two contiguous 16B LDS loads.
__device__ inline v16bf frag_a(const AS3 __bf16* S, int stride, int rbase, int k0,
                               int kcap, int lane) {
  int r = rbase + (lane & 15), g = lane >> 4;
  int c0 = k0 + 16*g, c1 = k0 + 16*g + 8;
  v8bf lo = (c0 < kcap) ? ldsv8(S + r*stride + c0) : zero8();
  v8bf hi = (c1 < kcap) ? ldsv8(S + r*stride + c1) : zero8();
  return combine(lo, hi);
}

// B fragment (32x16): lane holds column (lane&15); half-wave g covers K = k0+16g..+15
// -> two contiguous 16B LDS loads from row-major activation slab S[m][k].
__device__ inline v16bf frag_b(const AS3 __bf16* S, int stride, int k0, int kcap, int lane) {
  int m = lane & 15, g = lane >> 4;
  int c = k0 + 16*g;
  v8bf lo = (c     < kcap) ? ldsv8(S + m*stride + c)     : zero8();
  v8bf hi = (c + 8 < kcap) ? ldsv8(S + m*stride + c + 8) : zero8();
  return combine(lo, hi);
}

__device__ inline void make_b1(const AS3 __bf16* X, v16bf* bfr, int lane) {
#pragma unroll
  for (int kt = 0; kt < 4; ++kt) bfr[kt] = frag_b(X, HID, kt*32, 128, lane);
}
__device__ inline void make_b2(const AS3 __bf16* X1, const AS3 __bf16* X2, v16bf* bfr, int lane) {
  int m = lane & 15, g = lane >> 4;
#pragma unroll
  for (int kt = 0; kt < 4; ++kt) {
    int cb = kt*32 + 16*g;
    v8bf lo = add8(ldsv8(X1 + m*HID + cb),     ldsv8(X2 + m*HID + cb));
    v8bf hi = add8(ldsv8(X1 + m*HID + cb + 8), ldsv8(X2 + m*HID + cb + 8));
    bfr[kt] = combine(lo, hi);
  }
}
__device__ inline void make_b3(const AS3 __bf16* X1, const AS3 __bf16* X2,
                               const AS3 __bf16* X3, v16bf* bfr, int lane) {
  int m = lane & 15, g = lane >> 4;
#pragma unroll
  for (int kt = 0; kt < 4; ++kt) {
    int cb = kt*32 + 16*g;
    v8bf lo = add8_3(ldsv8(X1 + m*HID + cb),     ldsv8(X2 + m*HID + cb),     ldsv8(X3 + m*HID + cb));
    v8bf hi = add8_3(ldsv8(X1 + m*HID + cb + 8), ldsv8(X2 + m*HID + cb + 8), ldsv8(X3 + m*HID + cb + 8));
    bfr[kt] = combine(lo, hi);
  }
}

// Y^T = tanh(W^T @ X^T + b): 8 feature tiles, D-tile row = feature, col = sample.
__device__ inline void fwd_layer(const AS3 __bf16* WT, const AS3 float* bias,
                                 const v16bf* bfr, AS3 __bf16* Y, int lane) {
  int m = lane & 15, g = lane >> 4;
#pragma unroll
  for (int nt = 0; nt < 8; ++nt) {
    v8f cc = *(const AS3 v8f*)(bias + nt*16 + 8*g);
#pragma unroll
    for (int kt = 0; kt < 4; ++kt) {
      v16bf a = frag_a(WT, HID, nt*16, kt*32, 128, lane);
      cc = wmma_bf16(a, bfr[kt], cc);
    }
    v8bf o;
#pragma unroll
    for (int r = 0; r < 8; ++r) o[r] = (__bf16)fast_tanh(cc[r]);
    stsv8(Y + m*HID + nt*16 + 8*g, o);
  }
}

// acc = W @ G^T  (i.e. (G @ W^T)^T): backward matmul, W stored row-major.
__device__ inline void bwd_mat(const AS3 __bf16* W, const v16bf* bfr, v8f* acc, int lane) {
#pragma unroll
  for (int it = 0; it < 8; ++it) {
    v8f cc;
#pragma unroll
    for (int r = 0; r < 8; ++r) cc[r] = 0.0f;
#pragma unroll
    for (int kt = 0; kt < 4; ++kt) {
      v16bf a = frag_a(W, HID, it*16, kt*32, 128, lane);
      cc = wmma_bf16(a, bfr[kt], cc);
    }
    acc[it] = cc;
  }
}

// O = acc * (1 - T^2), contiguous v8bf read/modify/write per tile.
__device__ inline void dstore(const v8f* acc, const AS3 __bf16* T, AS3 __bf16* O, int lane) {
  int m = lane & 15, g = lane >> 4;
#pragma unroll
  for (int it = 0; it < 8; ++it) {
    v8bf tv = ldsv8(T + m*HID + it*16 + 8*g);
    v8bf o;
#pragma unroll
    for (int r = 0; r < 8; ++r) {
      float t = (float)tv[r];
      o[r] = (__bf16)(acc[it][r] * (1.0f - t*t));
    }
    stsv8(O + m*HID + it*16 + 8*g, o);
  }
}

// Forward + Hutchinson VJP. y/eps per-lane (lane&15 = sample m, duplicated across halves).
__device__ __noinline__ void aug_eval(const Ctx& c, float t, const float* yv,
                                      const float* epsv, float* kout, float* trout,
                                      int lane) {
  const int m = lane & 15, g = lane >> 4;
  // ---- stage input [t, y, 0-pad] row m (cols 0..15) ----
  if (g == 0) {
    v8bf s0, s1;
    s0[0] = (__bf16)t;
#pragma unroll
    for (int j = 0; j < 7; ++j) s0[j + 1] = (__bf16)yv[j];
    s1 = zero8();
    s1[0] = (__bf16)yv[7];
    stsv8(c.gA + m*HID,     s0);
    stsv8(c.gA + m*HID + 8, s1);
  }
  // ---- forward ----
  { // embedding: K padded to 16
    v16bf b = frag_b(c.gA, HID, 0, 16, lane);
#pragma unroll
    for (int nt = 0; nt < 8; ++nt) {
      v8f cc = *(const AS3 v8f*)(c.embB + nt*16 + 8*g);
      v16bf a = frag_a(c.embWT, 16, nt*16, 0, 16, lane);
      cc = wmma_bf16(a, b, cc);
      v8bf o;
#pragma unroll
      for (int r = 0; r < 8; ++r) o[r] = (__bf16)fast_tanh(cc[r]);
      stsv8(c.t1 + m*HID + nt*16 + 8*g, o);
    }
  }
  v16bf bfr[4];
  make_b1(c.t1, bfr, lane);             fwd_layer(c.wt,             c.bL,       bfr, c.t2, lane);
  make_b1(c.t2, bfr, lane);             fwd_layer(c.wt + 16384,     c.bL + 128, bfr, c.t3, lane);
  make_b2(c.t1, c.t3, bfr, lane);       fwd_layer(c.wt + 2*16384,   c.bL + 256, bfr, c.t4, lane);
  make_b1(c.t4, bfr, lane);             fwd_layer(c.wt + 3*16384,   c.bL + 384, bfr, c.t5, lane);
  // final: f^T = finW^T @ (t1+t3+t5)^T + finB  (single tile; valid features on g==0)
  make_b3(c.t1, c.t3, c.t5, bfr, lane);
  {
    v8f cc;
    if (g == 0) cc = *(const AS3 v8f*)c.finB;
    else {
#pragma unroll
      for (int r = 0; r < 8; ++r) cc[r] = 0.0f;
    }
#pragma unroll
    for (int kt = 0; kt < 4; ++kt) {
      v16bf a = frag_a(c.finWT, HID, 0, kt*32, 128, lane);
      cc = wmma_bf16(a, bfr[kt], cc);
    }
#pragma unroll
    for (int r = 0; r < 8; ++r) {
      float v = cc[r];
      float o = __shfl_xor(v, 16, 32);   // broadcast g==0 half to g==1
      kout[r] = g ? o : v;
    }
  }
  // ---- backward ----
  v8f gs[8], acc[8];
  { // g5^T = finW @ eps^T  (K padded to 16)
    v16bf be = frag_b(c.epsB, 16, 0, 16, lane);
#pragma unroll
    for (int it = 0; it < 8; ++it) {
      v8f cc;
#pragma unroll
      for (int r = 0; r < 8; ++r) cc[r] = 0.0f;
      v16bf a = frag_a(c.finWB, 16, it*16, 0, 16, lane);
      gs[it] = wmma_bf16(a, be, cc);
    }
  }
  dstore(gs, c.t5, c.gA, lane);                                   // gt5
  make_b1(c.gA, bfr, lane); bwd_mat(c.wb + 3*16384, bfr, acc, lane);
  dstore(acc, c.t4, c.gB, lane);                                  // gt4
  make_b1(c.gB, bfr, lane); bwd_mat(c.wb + 2*16384, bfr, acc, lane);
#pragma unroll
  for (int it = 0; it < 8; ++it) gs[it] = gs[it] + acc[it];       // gxm = g5 + branch
  dstore(gs, c.t3, c.gA, lane);                                   // gt3
  make_b1(c.gA, bfr, lane); bwd_mat(c.wb + 16384, bfr, acc, lane);
  dstore(acc, c.t2, c.gB, lane);                                  // gt2
  make_b1(c.gB, bfr, lane); bwd_mat(c.wb, bfr, acc, lane);
#pragma unroll
  for (int it = 0; it < 8; ++it) acc[it] = gs[it] + acc[it];      // gt1 = gxm + branch
  dstore(acc, c.t1, c.gA, lane);                                  // ga1
  // ginp^T = embW @ ga1^T (single tile, features 0..8 valid)
  make_b1(c.gA, bfr, lane);
  v8f gi;
#pragma unroll
  for (int r = 0; r < 8; ++r) gi[r] = 0.0f;
#pragma unroll
  for (int kt = 0; kt < 4; ++kt) {
    v16bf a = frag_a(c.embWB, HID, 0, kt*32, 128, lane);
    gi = wmma_bf16(a, bfr[kt], gi);
  }
  // trace: sum over features 1..8 of ginp * eps
  float p = 0.0f;
  if (g == 0) {
#pragma unroll
    for (int r = 1; r < 8; ++r) p += gi[r] * epsv[r - 1];   // features 1..7
  } else {
    p = gi[0] * epsv[7];                                    // feature 8
  }
  p += __shfl_xor(p, 16, 32);
  *trout = p;
}

__device__ inline void integrate(const Ctx& c, float* y, float& ld, const float* epsv, int lane) {
  float t = 1.0f;
  const float dt = -1.0f / (float)NSTEPS;
  float k1[8], k2[8], k3[8], k4[8], yt[8];
  float l1, l2, l3, l4;
  for (int s = 0; s < NSTEPS; ++s) {
    aug_eval(c, t, y, epsv, k1, &l1, lane);
#pragma unroll
    for (int r = 0; r < 8; ++r) yt[r] = y[r] + 0.5f*dt*k1[r];
    aug_eval(c, t + 0.5f*dt, yt, epsv, k2, &l2, lane);
#pragma unroll
    for (int r = 0; r < 8; ++r) yt[r] = y[r] + 0.5f*dt*k2[r];
    aug_eval(c, t + 0.5f*dt, yt, epsv, k3, &l3, lane);
#pragma unroll
    for (int r = 0; r < 8; ++r) yt[r] = y[r] + dt*k3[r];
    aug_eval(c, t + dt, yt, epsv, k4, &l4, lane);
#pragma unroll
    for (int r = 0; r < 8; ++r)
      y[r] += (dt/6.0f) * (k1[r] + 2.0f*k2[r] + 2.0f*k3[r] + k4[r]);
    ld += (dt/6.0f) * (l1 + 2.0f*l2 + 2.0f*l3 + l4);
    t += dt;
  }
}

__device__ inline void load_weights(const PSet& P, AS3 char* smem, int lane) {
  AS3 __bf16* wt = (AS3 __bf16*)(smem + OFF_WT);
  AS3 __bf16* wb = (AS3 __bf16*)(smem + OFF_WB);
  for (int l = 0; l < 4; ++l) {
    const float* s = P.p[W00 + 2*l];
    for (int i = lane; i < HID*HID; i += 32) wb[l*HID*HID + i] = (__bf16)s[i];
    for (int i = lane; i < HID*HID; i += 32) {
      int n = i >> 7, k = i & 127;
      wt[l*HID*HID + i] = (__bf16)s[k*HID + n];
    }
  }
  { // embW^T [128][16] (k>=9 zero) and embW [16][128] (rows>=9 zero)
    const float* s = P.p[EMBW];
    AS3 __bf16* d = (AS3 __bf16*)(smem + OFF_EMBWT);
    for (int i = lane; i < HID*16; i += 32) {
      int n = i >> 4, k = i & 15;
      d[i] = (k < DIMN + 1) ? (__bf16)s[k*HID + n] : (__bf16)0.0f;
    }
    AS3 __bf16* e = (AS3 __bf16*)(smem + OFF_EMBWB);
    for (int i = lane; i < 16*HID; i += 32) {
      int r = i >> 7, j = i & 127;
      e[i] = (r < DIMN + 1) ? (__bf16)s[r*HID + j] : (__bf16)0.0f;
    }
  }
  { // finW^T [16][128] (rows>=8 zero) and finW [128][16] (cols>=8 zero)
    const float* s = P.p[FINW];
    AS3 __bf16* d = (AS3 __bf16*)(smem + OFF_FINWT);
    for (int i = lane; i < 16*HID; i += 32) {
      int n = i >> 7, k = i & 127;
      d[i] = (n < DIMN) ? (__bf16)s[k*DIMN + n] : (__bf16)0.0f;
    }
    AS3 __bf16* e = (AS3 __bf16*)(smem + OFF_FINWB);
    for (int i = lane; i < HID*16; i += 32) {
      int r = i >> 4, j = i & 15;
      e[i] = (j < DIMN) ? (__bf16)s[r*DIMN + j] : (__bf16)0.0f;
    }
  }
  AS3 float* embB = (AS3 float*)(smem + OFF_EMBB);
  for (int i = lane; i < HID; i += 32) embB[i] = P.p[EMBB][i];
  AS3 float* bL = (AS3 float*)(smem + OFF_BL);
  for (int l = 0; l < 4; ++l)
    for (int i = lane; i < HID; i += 32) bL[l*HID + i] = P.p[B00 + 2*l][i];
  AS3 float* finB = (AS3 float*)(smem + OFF_FINB);
  for (int i = lane; i < DIMN; i += 32) finB[i] = P.p[FINB][i];
}

__global__ __launch_bounds__(32)
void ffjord_nvp_kernel(PSet P0, PSet P1, const float* __restrict__ lu,
                       const float* __restrict__ x, const int* __restrict__ perm,
                       const int* __restrict__ epsI, float* __restrict__ out) {
  extern __shared__ char smemRaw[];
  AS3 char* smem = (AS3 char*)smemRaw;
  const int lane = threadIdx.x & 31;
  const int m = lane & 15, g = lane >> 4;
  const int row = blockIdx.x*16 + m;

  Ctx c;
  c.wt    = (const AS3 __bf16*)(smem + OFF_WT);
  c.wb    = (const AS3 __bf16*)(smem + OFF_WB);
  c.embWT = (const AS3 __bf16*)(smem + OFF_EMBWT);
  c.embWB = (const AS3 __bf16*)(smem + OFF_EMBWB);
  c.finWT = (const AS3 __bf16*)(smem + OFF_FINWT);
  c.finWB = (const AS3 __bf16*)(smem + OFF_FINWB);
  c.embB  = (const AS3 float*)(smem + OFF_EMBB);
  c.bL    = (const AS3 float*)(smem + OFF_BL);
  c.finB  = (const AS3 float*)(smem + OFF_FINB);
  c.t1 = (AS3 __bf16*)(smem + OFF_T1);
  c.t2 = c.t1 + 16*HID; c.t3 = c.t2 + 16*HID; c.t4 = c.t3 + 16*HID; c.t5 = c.t4 + 16*HID;
  c.gA = (AS3 __bf16*)(smem + OFF_GA);
  c.gB = (AS3 __bf16*)(smem + OFF_GB);
  c.epsB = (AS3 __bf16*)(smem + OFF_EPSB);

  // ---------- phase 0 ----------
  load_weights(P0, smem, lane);
  float epsv[8], y[8];
  float ld = 0.0f;
#pragma unroll
  for (int i = 0; i < 8; ++i) {
    epsv[i] = (float)(2*epsI[row*DIMN + i] - 1);
    y[i]    = x[row*DIMN + i];
  }
  if (g == 0) {
    v8bf e;
#pragma unroll
    for (int j = 0; j < 8; ++j) e[j] = (__bf16)epsv[j];
    stsv8(c.epsB + m*16,     e);
    stsv8(c.epsB + m*16 + 8, zero8());
  }
  integrate(c, y, ld, epsv, lane);

  // ---------- LU solve (per lane, duplicated across halves): z = U^-1 L^-1 P^-1 y ----------
  {
    float xp[8], w[8], z[8];
#pragma unroll
    for (int i = 0; i < 8; ++i) xp[perm[i]] = y[i];
#pragma unroll
    for (int i = 0; i < 8; ++i) {           // unit-diag L forward-sub
      float s = xp[i];
      for (int j = 0; j < i; ++j) s -= lu[i*8 + j] * w[j];
      w[i] = s;
    }
#pragma unroll
    for (int i = 7; i >= 0; --i) {          // U back-sub
      float s = w[i];
      for (int j = i + 1; j < 8; ++j) s -= lu[i*8 + j] * z[j];
      z[i] = s / lu[i*8 + i];
    }
#pragma unroll
    for (int i = 0; i < 8; ++i) y[i] = z[i];
  }

  // ---------- phase 1 ----------
  load_weights(P1, smem, lane);
#pragma unroll
  for (int i = 0; i < 8; ++i)
    epsv[i] = (float)(2*epsI[BATCHN*DIMN + row*DIMN + i] - 1);
  if (g == 0) {
    v8bf e;
#pragma unroll
    for (int j = 0; j < 8; ++j) e[j] = (__bf16)epsv[j];
    stsv8(c.epsB + m*16,     e);
    stsv8(c.epsB + m*16 + 8, zero8());
  }
  integrate(c, y, ld, epsv, lane);

  // ---------- finalize ----------
  float ild = 0.0f;
#pragma unroll
  for (int i = 0; i < 8; ++i) ild -= logf(fabsf(lu[i*8 + i]));
  if (g == 0) {
    float s = 0.0f;
#pragma unroll
    for (int i = 0; i < 8; ++i) s += y[i]*y[i] + LOG2PI_F;
    out[row] = -0.5f*s + ld + ild;
  }
}

extern "C" void kernel_launch(void* const* d_in, const int* in_sizes, int n_in,
                              void* d_out, int out_size, void* d_ws, size_t ws_size,
                              hipStream_t stream) {
  (void)d_ws; (void)ws_size; (void)out_size;
  int base0 = 0, base1 = 12, luI = 24, xI = 25, permI = 26, epsI = 27;
  if (n_in >= 28 && in_sizes[0] == 2*BATCHN*DIMN) {
    // outer dict alphabetized: eps_int, lu, params0(12), params1(12), perm, x
    epsI = 0; luI = 1; base0 = 2; base1 = 14; permI = 26; xI = 27;
  }
  auto mk = [&](int base) {
    PSet P;
    if (in_sizes[base] == (DIMN + 1)*HID) {
      // insertion order: emb_w, emb_b, b0_w0, b0_b0, b0_w1, b0_b1, b1_w0, b1_b0, b1_w1, b1_b1, fin_w, fin_b
      P.p[EMBW] = (const float*)d_in[base+0];  P.p[EMBB] = (const float*)d_in[base+1];
      P.p[W00]  = (const float*)d_in[base+2];  P.p[B00]  = (const float*)d_in[base+3];
      P.p[W01]  = (const float*)d_in[base+4];  P.p[B01]  = (const float*)d_in[base+5];
      P.p[W10]  = (const float*)d_in[base+6];  P.p[B10]  = (const float*)d_in[base+7];
      P.p[W11]  = (const float*)d_in[base+8];  P.p[B11]  = (const float*)d_in[base+9];
      P.p[FINW] = (const float*)d_in[base+10]; P.p[FINB] = (const float*)d_in[base+11];
    } else {
      // alphabetical: b0_b0, b0_b1, b0_w0, b0_w1, b1_b0, b1_b1, b1_w0, b1_w1, emb_b, emb_w, fin_b, fin_w
      P.p[B00]  = (const float*)d_in[base+0];  P.p[B01]  = (const float*)d_in[base+1];
      P.p[W00]  = (const float*)d_in[base+2];  P.p[W01]  = (const float*)d_in[base+3];
      P.p[B10]  = (const float*)d_in[base+4];  P.p[B11]  = (const float*)d_in[base+5];
      P.p[W10]  = (const float*)d_in[base+6];  P.p[W11]  = (const float*)d_in[base+7];
      P.p[EMBB] = (const float*)d_in[base+8];  P.p[EMBW] = (const float*)d_in[base+9];
      P.p[FINB] = (const float*)d_in[base+10]; P.p[FINW] = (const float*)d_in[base+11];
    }
    return P;
  };
  PSet P0 = mk(base0), P1 = mk(base1);
  ffjord_nvp_kernel<<<BATCHN/16, 32, LDS_BYTES, stream>>>(
      P0, P1,
      (const float*)d_in[luI], (const float*)d_in[xI],
      (const int*)d_in[permI], (const int*)d_in[epsI],
      (float*)d_out);
}
